// Attention_83330955478086
// MI455X (gfx1250) — matrix operand order repaired and checked by reading, not compiled, not run on hardware
//
#include <hip/hip_runtime.h>
#include <hip/hip_bf16.h>

#define DIM 4096
#define NH  32
#define HD  128
#define SEQ 2048

typedef __attribute__((ext_vector_type(16))) __bf16 v16bf;
typedef __attribute__((ext_vector_type(8)))  float  v8f;

union FragAB { v16bf v; int4 q[2]; __bf16 h[16]; };
union FragC  { v8f   v; float f[8]; };

#define GAS __attribute__((address_space(1)))
#define LAS __attribute__((address_space(3)))

// generic (gcc-style) vector to exactly match the builtin's V4i pointee type
typedef int v4i_ __attribute__((vector_size(16)));

// ---- CDNA5 async global->LDS copy path (ASYNCcnt), with safe fallback ----
#if __has_builtin(__builtin_amdgcn_global_load_async_to_lds_b128)
#define HAVE_ASYNC 1
static __device__ __forceinline__ void async_cp16(void* lds, const void* g) {
  __builtin_amdgcn_global_load_async_to_lds_b128((GAS v4i_*)g, (LAS v4i_*)lds, 0, 0);
}
#define WAIT_ASYNC(n) asm volatile("s_wait_asynccnt " #n ::: "memory")
#else
#define HAVE_ASYNC 0
static __device__ __forceinline__ void async_cp16(void* lds, const void* g) {
  *(int4*)lds = *(const int4*)g;
}
#define WAIT_ASYNC(n) do {} while (0)
#endif

static __device__ __forceinline__ v8f wmma_bf16(v16bf a, v16bf b, v8f c) {
  // D = A(16x32 bf16) * B(32x16 bf16) + C(16x16 f32)
  return __builtin_amdgcn_wmma_f32_16x16x32_bf16(false, a, false, b, (short)0, c, false, false);
}

static __device__ __forceinline__ float hmax16(float v) {
  #pragma unroll
  for (int off = 8; off > 0; off >>= 1) v = fmaxf(v, __shfl_xor(v, off, 16));
  return v;
}
static __device__ __forceinline__ float hsum16(float v) {
  #pragma unroll
  for (int off = 8; off > 0; off >>= 1) v += __shfl_xor(v, off, 16);
  return v;
}

// ---------------------------------------------------------------------------
// Kernel 1: fp32 -> bf16 convert (x)
// ---------------------------------------------------------------------------
__global__ __launch_bounds__(256) void cvt_f32_bf16(const float* __restrict__ x,
                                                    __bf16* __restrict__ y) {
  size_t i = ((size_t)blockIdx.x * 256 + threadIdx.x) * 8;
  float4 a = *(const float4*)(x + i);
  float4 b = *(const float4*)(x + i + 4);
  union { __bf16 h[8]; int4 q; } u;
  u.h[0] = (__bf16)a.x; u.h[1] = (__bf16)a.y; u.h[2] = (__bf16)a.z; u.h[3] = (__bf16)a.w;
  u.h[4] = (__bf16)b.x; u.h[5] = (__bf16)b.y; u.h[6] = (__bf16)b.z; u.h[7] = (__bf16)b.w;
  *(int4*)(y + i) = u.q;
}

// ---------------------------------------------------------------------------
// Shared GEMM core: O = X(bf16 [M][4096]) @ W(fp32 [N][4096]).T
// Block = 128 threads (4 waves). Block tile 64x64, wave tile 32x32, K step 32.
// Double-buffered LDS; X tile staged via async-to-LDS (ASYNCcnt), W tile
// converted fp32->bf16 in VGPRs (co-executes with WMMA).
// ---------------------------------------------------------------------------
static __device__ __forceinline__ void gemm_core(const __bf16* __restrict__ X,
                                                 const float*  __restrict__ W,
                                                 int m0, int n0, int lane, int wave,
                                                 FragC acc[2][2]) {
  __shared__ __align__(16) __bf16 sX[2][64][32];
  __shared__ __align__(16) __bf16 sW[2][64][32];
  const int tid = wave * 32 + lane;
  const int xr  = tid >> 1;           // tile row loaded by this thread
  const int xc  = (tid & 1) * 16;     // element offset within row
  const int wm  = (wave >> 1) * 32;   // wave sub-tile
  const int wn  = (wave & 1) * 32;

  auto stage = [&](int buf, int k0) {
    const __bf16* xp = X + (size_t)(m0 + xr) * DIM + k0 + xc;
    async_cp16(&sX[buf][xr][xc],     xp);
    async_cp16(&sX[buf][xr][xc + 8], xp + 8);
    const float* wp = W + (size_t)(n0 + xr) * DIM + k0 + xc;
    float4 w0 = *(const float4*)(wp + 0);
    float4 w1 = *(const float4*)(wp + 4);
    float4 w2 = *(const float4*)(wp + 8);
    float4 w3 = *(const float4*)(wp + 12);
    union { __bf16 h[16]; int4 q[2]; } uw;
    uw.h[0]=(__bf16)w0.x;  uw.h[1]=(__bf16)w0.y;  uw.h[2]=(__bf16)w0.z;  uw.h[3]=(__bf16)w0.w;
    uw.h[4]=(__bf16)w1.x;  uw.h[5]=(__bf16)w1.y;  uw.h[6]=(__bf16)w1.z;  uw.h[7]=(__bf16)w1.w;
    uw.h[8]=(__bf16)w2.x;  uw.h[9]=(__bf16)w2.y;  uw.h[10]=(__bf16)w2.z; uw.h[11]=(__bf16)w2.w;
    uw.h[12]=(__bf16)w3.x; uw.h[13]=(__bf16)w3.y; uw.h[14]=(__bf16)w3.z; uw.h[15]=(__bf16)w3.w;
    *(int4*)&sW[buf][xr][xc]     = uw.q[0];
    *(int4*)&sW[buf][xr][xc + 8] = uw.q[1];
  };

  stage(0, 0);
  int cur = 0;
  for (int k0 = 0; k0 < DIM; k0 += 32, cur ^= 1) {
    if (k0 + 32 < DIM) {
      __builtin_prefetch(W + (size_t)(n0 + xr) * DIM + k0 + 64 + xc, 0, 1);
      stage(cur ^ 1, k0 + 32);
      WAIT_ASYNC(2);       // previous buffer's async copies complete
    } else {
      WAIT_ASYNC(0);
    }
    __syncthreads();       // all waves' copies for `cur` visible

    FragAB afr[2], bfr[2];
    #pragma unroll
    for (int fi = 0; fi < 2; ++fi) {
      const int m  = wm + fi * 16 + (lane & 15);
      const int ko = (lane >> 4) * 16;         // A: lanes 0-15 K 0..15; 16-31 K 16..31
      afr[fi].q[0] = *(const int4*)&sX[cur][m][ko];
      afr[fi].q[1] = *(const int4*)&sX[cur][m][ko + 8];
    }
    #pragma unroll
    for (int fj = 0; fj < 2; ++fj) {
      const int n  = wn + fj * 16 + (lane & 15);
      const int kk = (lane >> 4) * 16;         // B: lanes 0-15 K 0..15; 16-31 K 16..31
      bfr[fj].q[0] = *(const int4*)&sW[cur][n][kk];
      bfr[fj].q[1] = *(const int4*)&sW[cur][n][kk + 8];
    }
    #pragma unroll
    for (int fi = 0; fi < 2; ++fi)
      #pragma unroll
      for (int fj = 0; fj < 2; ++fj)
        acc[fi][fj].v = wmma_bf16(afr[fi].v, bfr[fj].v, acc[fi][fj].v);
    __syncthreads();       // readers done before next iteration overwrites
  }
}

// ---------------------------------------------------------------------------
// Kernel 2: QKV projection + fused "rotary" scale (reference rotary is
// elementwise: x * (cos -/+ sin)), scatter to attention layouts:
//   Q,K -> [h][pos][d] bf16 ; V -> transposed [h][d][pos] bf16
// grid = (SEQ/64, DIM/64, 3)
// ---------------------------------------------------------------------------
__global__ __launch_bounds__(128) void gemm_qkv(const __bf16* __restrict__ X,
                                                const float* __restrict__ Wq,
                                                const float* __restrict__ Wk,
                                                const float* __restrict__ Wv,
                                                const float* __restrict__ freqs,
                                                __bf16* __restrict__ Qo,
                                                __bf16* __restrict__ Ko,
                                                __bf16* __restrict__ Vt) {
  const int lane = threadIdx.x & 31, wave = threadIdx.x >> 5;
  const int m0 = blockIdx.x * 64, n0 = blockIdx.y * 64;
  const int z = blockIdx.z;
  const float* W = (z == 0) ? Wq : (z == 1) ? Wk : Wv;

  FragC acc[2][2];
  #pragma unroll
  for (int i = 0; i < 2; ++i)
    #pragma unroll
    for (int j = 0; j < 2; ++j)
      #pragma unroll
      for (int v = 0; v < 8; ++v) acc[i][j].f[v] = 0.f;

  gemm_core(X, W, m0, n0, lane, wave, acc);

  const int wm = (wave >> 1) * 32, wn = (wave & 1) * 32;
  #pragma unroll
  for (int fi = 0; fi < 2; ++fi)
    #pragma unroll
    for (int fj = 0; fj < 2; ++fj)
      #pragma unroll
      for (int v = 0; v < 8; ++v) {
        const int m = m0 + wm + fi * 16 + v + ((lane >> 4) << 3);
        const int n = n0 + wn + fj * 16 + (lane & 15);
        float val = acc[fi][fj].f[v];
        const int hh = n >> 7, d = n & 127;
        if (z < 2) {
          const float c = freqs[((size_t)m * 64 + (d >> 1)) * 2];
          const float s = freqs[((size_t)m * 64 + (d >> 1)) * 2 + 1];
          val *= c + ((d & 1) ? s : -s);
          const __bf16 bv = (__bf16)val;
          if (z == 0) Qo[((size_t)hh * SEQ + m) * HD + d] = bv;
          else        Ko[((size_t)hh * SEQ + m) * HD + d] = bv;
        } else {
          Vt[((size_t)hh * HD + d) * SEQ + m] = (__bf16)val;  // transposed V
        }
      }
}

// ---------------------------------------------------------------------------
// Kernel 3: flash attention. Block = 128 threads (4 waves), 64 queries/block,
// one head per blockIdx.y. K (32x128) and V^T (128x32) tiles staged into LDS
// via async copies and shared by all 4 waves (4x L2 traffic reduction).
// Online softmax kept in the WMMA C layout; P re-fragmented via per-wave LDS.
// Waves past their causal limit compute fully-masked (exp->0) blocks: no-op.
// grid = (SEQ/64, NH)
// ---------------------------------------------------------------------------
__global__ __launch_bounds__(128) void flash_attn(const __bf16* __restrict__ Q,
                                                  const __bf16* __restrict__ K,
                                                  const __bf16* __restrict__ Vt,
                                                  __bf16* __restrict__ Attn) {
  __shared__ __align__(16) __bf16 sK[32][128];     // [key][d]
  __shared__ __align__(16) __bf16 sV[128][32];     // [d][key]
  __shared__ __align__(16) __bf16 sP[4][16][32];   // per-wave P tile
  const int tid  = threadIdx.x;
  const int lane = tid & 31, wave = tid >> 5;
  const int qb0  = blockIdx.x * 64;
  const int qb   = qb0 + wave * 16;
  const int h    = blockIdx.y;
  const __bf16* Qh = Q  + (size_t)h * SEQ * HD;
  const __bf16* Kh = K  + (size_t)h * SEQ * HD;
  const __bf16* Vh = Vt + (size_t)h * HD * SEQ;

  // Q A-fragments over head dim: 4 x (16x32), direct from global (read once)
  FragAB a[4];
  {
    const int qm = qb + (lane & 15);
    const int ko = (lane >> 4) * 16;
    #pragma unroll
    for (int f = 0; f < 4; ++f) {
      const __bf16* p = Qh + (size_t)qm * HD + f * 32 + ko;
      a[f].q[0] = *(const int4*)p;
      a[f].q[1] = *(const int4*)(p + 8);
    }
  }

  FragC o[8];
  float mrow[8], lrow[8];
  #pragma unroll
  for (int f = 0; f < 8; ++f)
    #pragma unroll
    for (int v = 0; v < 8; ++v) o[f].f[v] = 0.f;
  #pragma unroll
  for (int v = 0; v < 8; ++v) { mrow[v] = -3.0e38f; lrow[v] = 0.f; }

  const float scale = 0.08838834764831845f;  // 1/sqrt(128)
  const int kend = qb0 + 64;                 // block-uniform causal bound

  for (int kb = 0; kb < kend; kb += 32) {
    // --- stage K and V tiles via async copies (8KB + 8KB) ---
    #pragma unroll
    for (int j = 0; j < 4; ++j) {
      const int c  = tid * 4 + j;       // 0..511
      const int kr = c >> 4;            // 0..31   (K rows)
      const int kc = (c & 15) * 8;      // 0..120
      async_cp16(&sK[kr][kc], Kh + (size_t)(kb + kr) * HD + kc);
      const int dr = c >> 2;            // 0..127  (V rows = head dim)
      const int dc = (c & 3) * 8;       // 0..24
      async_cp16(&sV[dr][dc], Vh + (size_t)dr * SEQ + kb + dc);
    }
    WAIT_ASYNC(0);
    __syncthreads();

    // --- S = Q K^T for 32 keys, two 16x16 C frags ---
    FragC s0, s1;
    #pragma unroll
    for (int v = 0; v < 8; ++v) { s0.f[v] = 0.f; s1.f[v] = 0.f; }
    const int nrow = lane & 15;
    const int key0 = kb + nrow;
    #pragma unroll
    for (int f = 0; f < 4; ++f) {
      const int doff = f * 32 + (lane >> 4) * 16;
      FragAB b0, b1;
      b0.q[0] = *(const int4*)&sK[nrow][doff];
      b0.q[1] = *(const int4*)&sK[nrow][doff + 8];
      b1.q[0] = *(const int4*)&sK[nrow + 16][doff];
      b1.q[1] = *(const int4*)&sK[nrow + 16][doff + 8];
      s0.v = wmma_bf16(a[f].v, b0.v, s0.v);
      s1.v = wmma_bf16(a[f].v, b1.v, s1.v);
    }

    // --- mask + scale + online softmax (row = v + 8*(lane>=16)) ---
    #pragma unroll
    for (int v = 0; v < 8; ++v) {
      const int qrow = qb + v + ((lane >> 4) << 3);
      float v0 = s0.f[v] * scale + ((key0      > qrow) ? -1.0e9f : 0.f);
      float v1 = s1.f[v] * scale + ((key0 + 16 > qrow) ? -1.0e9f : 0.f);
      const float rmax = hmax16(fmaxf(v0, v1));
      const float newm = fmaxf(mrow[v], rmax);
      const float alpha = __expf(mrow[v] - newm);
      mrow[v] = newm;
      const float p0 = __expf(v0 - newm);
      const float p1 = __expf(v1 - newm);
      lrow[v] = lrow[v] * alpha + hsum16(p0 + p1);
      #pragma unroll
      for (int f = 0; f < 8; ++f) o[f].f[v] *= alpha;
      const int pr = v + ((lane >> 4) << 3);
      sP[wave][pr][nrow]      = (__bf16)p0;
      sP[wave][pr][nrow + 16] = (__bf16)p1;
    }
    asm volatile("" ::: "memory");  // per-wave LDS is in-order; block reordering only

    // --- P as A fragment (16x32) ---
    FragAB pf;
    {
      const __bf16* pp = &sP[wave][lane & 15][(lane >> 4) * 16];
      pf.q[0] = *(const int4*)pp;
      pf.q[1] = *(const int4*)(pp + 8);
    }
    // --- O += P @ V (V already transposed: contiguous B-fragments) ---
    #pragma unroll
    for (int f = 0; f < 8; ++f) {
      const int d = f * 16 + (lane & 15);
      FragAB bv;
      bv.q[0] = *(const int4*)&sV[d][(lane >> 4) * 16];
      bv.q[1] = *(const int4*)&sV[d][(lane >> 4) * 16 + 8];
      o[f].v = wmma_bf16(pf.v, bv.v, o[f].v);
    }
    __syncthreads();  // readers done before next tile staging overwrites
  }

  // normalize + write attn output as bf16 [pos][h*128+d]
  float inv[8];
  #pragma unroll
  for (int v = 0; v < 8; ++v) inv[v] = 1.0f / lrow[v];
  #pragma unroll
  for (int f = 0; f < 8; ++f)
    #pragma unroll
    for (int v = 0; v < 8; ++v) {
      const int m = qb + v + ((lane >> 4) << 3);
      const int n = h * HD + f * 16 + (lane & 15);
      Attn[(size_t)m * DIM + n] = (__bf16)(o[f].f[v] * inv[v]);
    }
}

// ---------------------------------------------------------------------------
// Kernel 4: output projection, fp32 result to d_out. grid = (SEQ/64, DIM/64)
// ---------------------------------------------------------------------------
__global__ __launch_bounds__(128) void gemm_out(const __bf16* __restrict__ X,
                                                const float* __restrict__ W,
                                                float* __restrict__ Out) {
  const int lane = threadIdx.x & 31, wave = threadIdx.x >> 5;
  const int m0 = blockIdx.x * 64, n0 = blockIdx.y * 64;

  FragC acc[2][2];
  #pragma unroll
  for (int i = 0; i < 2; ++i)
    #pragma unroll
    for (int j = 0; j < 2; ++j)
      #pragma unroll
      for (int v = 0; v < 8; ++v) acc[i][j].f[v] = 0.f;

  gemm_core(X, W, m0, n0, lane, wave, acc);

  const int wm = (wave >> 1) * 32, wn = (wave & 1) * 32;
  #pragma unroll
  for (int fi = 0; fi < 2; ++fi)
    #pragma unroll
    for (int fj = 0; fj < 2; ++fj)
      #pragma unroll
      for (int v = 0; v < 8; ++v) {
        const int m = m0 + wm + fi * 16 + v + ((lane >> 4) << 3);
        const int n = n0 + wn + fj * 16 + (lane & 15);
        Out[(size_t)m * DIM + n] = acc[fi][fj].f[v];
      }
}

// ---------------------------------------------------------------------------
// Launch
// ---------------------------------------------------------------------------
extern "C" void kernel_launch(void* const* d_in, const int* in_sizes, int n_in,
                              void* d_out, int out_size, void* d_ws, size_t ws_size,
                              hipStream_t stream) {
  const float* x     = (const float*)d_in[0];
  // d_in[1] = start_pos (0), d_in[3] = mask (causal applied inline)
  const float* freqs = (const float*)d_in[2];
  const float* wq    = (const float*)d_in[4];
  const float* wk    = (const float*)d_in[5];
  const float* wv    = (const float*)d_in[6];
  const float* wo    = (const float*)d_in[7];
  float* out = (float*)d_out;

  const size_t BUF = (size_t)SEQ * DIM * sizeof(__bf16);  // 16 MB each
  char* w = (char*)d_ws;
  __bf16* xb = (__bf16*)(w);
  __bf16* qb = (__bf16*)(w + BUF);
  __bf16* kb = (__bf16*)(w + 2 * BUF);
  __bf16* vt = (__bf16*)(w + 3 * BUF);
  __bf16* at = (__bf16*)(w + 4 * BUF);

  cvt_f32_bf16<<<(SEQ * DIM) / (256 * 8), 256, 0, stream>>>(x, xb);
  gemm_qkv<<<dim3(SEQ / 64, DIM / 64, 3), 128, 0, stream>>>(xb, wq, wk, wv, freqs, qb, kb, vt);
  flash_attn<<<dim3(SEQ / 64, NH), 128, 0, stream>>>(qb, kb, vt, at);
  gemm_out<<<dim3(SEQ / 64, DIM / 64), 128, 0, stream>>>(at, wo, out);
}
